// LinformerSelfAttention_69827578298927
// MI455X (gfx1250) — matrix operand order repaired and checked by reading, not compiled, not run on hardware
//
#include <hip/hip_runtime.h>

typedef __attribute__((ext_vector_type(16))) __bf16 v16bf;
typedef __attribute__((ext_vector_type(8)))  float  v8f;

// --------------------------------------------------------- async-LDS probe --
#if defined(__has_builtin)
#if __has_builtin(__builtin_amdgcn_global_load_async_to_lds_b64) && \
    __has_builtin(__builtin_amdgcn_s_wait_asynccnt)
#define USE_ASYNC 1
#endif
#endif
#ifndef USE_ASYNC
#define USE_ASYNC 0
#endif

#if USE_ASYNC
// Builtin signature (from hipcc diagnostic): b64 variant takes
// (v2i addrspace(1)*, v2i addrspace(3)*, imm offset, imm cpol),
// where v2i is a GCC-vector "int __vector(2)".
typedef int async_v2i __attribute__((vector_size(8)));
// integer round-trips: AS1 generic pointers are bit-identical; AS3 offset ==
// low 32 bits of the generic address (aperture rule).
#define AS1P(p) ((__attribute__((address_space(1))) async_v2i*)(unsigned long long)(p))
#define AS3P(p) ((__attribute__((address_space(3))) async_v2i*)(unsigned int)(unsigned long long)(p))
#endif

// ---------------------------------------------------------------- helpers ---
__device__ __forceinline__ unsigned short f2bf(float f) {
  unsigned int u = __float_as_uint(f);
  u += 0x7FFFu + ((u >> 16) & 1u);          // round-to-nearest-even
  return (unsigned short)(u >> 16);
}

union FragBF {          // 16 bf16 = 32 bytes = two 16B LDS chunks
  uint4 q[2];
  v16bf v;
};

#define TILE_M 128
#define TILE_N 64
#define TILE_K 32
#define LDSP   40   // padded row stride in halves (80B: keeps 16B chunk alignment)

// --------------------------------------------------- generic bf16 WMMA GEMM -
// C[M,N] = alpha * A[M,K] @ B + (bias)    A row-major bf16 (lda)
// TRANSB=0: B row-major [K,N] (ldb) ; TRANSB=1: B given as [N,K] rows (ldb)
// batch z: offset = (z/batchH)*s?o + (z%batchH)*s?i  (elements)
// Double-buffered LDS; tile kt+1 staged via GLOBAL_LOAD_ASYNC_TO_LDS_B64
// (ASYNCcnt) while tile kt feeds v_wmma_f32_16x16x32_bf16.
template<bool TRANSB, bool OUTBF16, bool BIAS>
__global__ __launch_bounds__(256)
void gemm_bf16_wmma(const unsigned short* __restrict__ A,
                    const unsigned short* __restrict__ B,
                    void* __restrict__ Cout,
                    int M, int N, int K,
                    int lda, int ldb, int ldc,
                    int batchH,
                    long long sAo, long long sAi,
                    long long sBo, long long sBi,
                    long long sCo, long long sCi,
                    float alpha,
                    const float* __restrict__ bias)
{
  __shared__ unsigned short Asl[2][TILE_M * LDSP];
  __shared__ unsigned short Bsl[2][TILE_N * LDSP];

  const int tid   = threadIdx.x;
  const int lane  = tid & 31;
  const int wave  = tid >> 5;
  const int wm    = wave & 3;          // 4 waves along M (4*32 = 128)
  const int wn    = wave >> 2;         // 2 waves along N (2*32 = 64)
  const int z     = blockIdx.z;
  const int zo    = z / batchH;
  const int zi    = z - zo * batchH;

  const unsigned short* Ab = A + zo * sAo + zi * sAi;
  const unsigned short* Bb = B + zo * sBo + zi * sBi;

  const int m0 = blockIdx.y * TILE_M;
  const int n0 = blockIdx.x * TILE_N;

  const int mrow  = lane & 15;         // M (A) / N (B) index within 16
  const int khalf = lane >> 4;         // K-half select per ISA layout

  // async ops issued per tile by each wave (A: 4 x b64; TRANSB B: 2 x b64)
  constexpr int PEND = TRANSB ? 6 : 4;

  // ---- tile staging: global -> LDS buffer `buf` for k-block `kt` ----
  auto stage = [&](int kt, int buf) {
    const int k0 = kt * TILE_K;
    {   // A tile [128 x 32]
      const unsigned short* ag = Ab + (long long)m0 * lda + k0;
      const int r = tid >> 3;              // 0..31
      const int c = (tid & 7) * 4;         // 0..28 (halves) -> 8B chunks
      #pragma unroll
      for (int p = 0; p < 4; ++p) {
        const int row = p * 32 + r;
        const unsigned short* gp = ag + (long long)row * lda + c;
        unsigned short* lp = &Asl[buf][row * LDSP + c];
#if USE_ASYNC
        __builtin_amdgcn_global_load_async_to_lds_b64(AS1P(gp), AS3P(lp), 0, 0);
#else
        *(uint2*)lp = *(const uint2*)gp;
#endif
      }
    }
    if (TRANSB) {
      // source rows are already output columns: Bsrc[n][k] -> direct copy
      const unsigned short* bg = Bb + (long long)n0 * ldb + k0;
      const int r = tid >> 3;
      const int c = (tid & 7) * 4;
      #pragma unroll
      for (int p = 0; p < 2; ++p) {
        const int row = p * 32 + r;
        const unsigned short* gp = bg + (long long)row * ldb + c;
        unsigned short* lp = &Bsl[buf][row * LDSP + c];
#if USE_ASYNC
        __builtin_amdgcn_global_load_async_to_lds_b64(AS1P(gp), AS3P(lp), 0, 0);
#else
        *(uint2*)lp = *(const uint2*)gp;
#endif
      }
    } else {
      // Bsrc[k][n] row-major: transpose 2x2 bf16 blocks in registers,
      // store as 32-bit words (k-pairs contiguous per output column n).
      const unsigned short* bg = Bb + (long long)k0 * ldb + n0;
      #pragma unroll
      for (int p = 0; p < 2; ++p) {
        const int cell = p * 256 + tid;        // 0..511
        const int k = (cell & 15) * 2;         // 0..30
        const int n = (cell >> 4) * 2;         // 0..62
        const unsigned int r0 = *(const unsigned int*)(bg + (long long)k * ldb + n);
        const unsigned int r1 = *(const unsigned int*)(bg + (long long)(k + 1) * ldb + n);
        const unsigned int w0 = (r0 & 0xFFFFu) | (r1 << 16);          // column n
        const unsigned int w1 = (r0 >> 16)     | (r1 & 0xFFFF0000u);  // column n+1
        *(unsigned int*)&Bsl[buf][n * LDSP + k]       = w0;
        *(unsigned int*)&Bsl[buf][(n + 1) * LDSP + k] = w1;
      }
    }
  };

  v8f zero;
  #pragma unroll
  for (int i = 0; i < 8; ++i) zero[i] = 0.0f;
  v8f acc[2][2] = {{zero, zero}, {zero, zero}};

  const int nK = K / TILE_K;
  stage(0, 0);                                   // prologue

  for (int kt = 0; kt < nK; ++kt) {
    const int cur = kt & 1;
    const bool more = (kt + 1 < nK);
    if (more) stage(kt + 1, cur ^ 1);            // overlap next-tile DMA

#if USE_ASYNC
    if (more) __builtin_amdgcn_s_wait_asynccnt(PEND);  // tile kt drained,
    else      __builtin_amdgcn_s_wait_asynccnt(0);     // tile kt+1 still flying
#endif
    __syncthreads();

    // ---- fragment loads per ISA VGPR layouts ----
    FragBF fa[2], fb[2];
    #pragma unroll
    for (int mi = 0; mi < 2; ++mi) {
      const unsigned short* ap =
          &Asl[cur][(wm * 32 + mi * 16 + mrow) * LDSP + khalf * 8];
      fa[mi].q[0] = *(const uint4*)(ap);        // K = kh*8 .. +8
      fa[mi].q[1] = *(const uint4*)(ap + 16);   // K = 16+kh*8 .. +8
    }
    #pragma unroll
    for (int ni = 0; ni < 2; ++ni) {
      const unsigned short* bp =
          &Bsl[cur][(wn * 32 + ni * 16 + mrow) * LDSP + khalf * 8];
      fb[ni].q[0] = *(const uint4*)(bp);        // K = kh*8 .. +8
      fb[ni].q[1] = *(const uint4*)(bp + 16);   // K = 16+kh*8 .. +8
    }

    #pragma unroll
    for (int mi = 0; mi < 2; ++mi)
      #pragma unroll
      for (int ni = 0; ni < 2; ++ni)
        acc[mi][ni] = __builtin_amdgcn_wmma_f32_16x16x32_bf16(
            false, fa[mi].v, false, fb[ni].v,
            (short)0, acc[mi][ni], false, false);

    __syncthreads();   // readers done before buffer `cur` is restaged (kt+2)
  }

  // ---- store C per 16x16 f32 D layout ----
  const long long cbase = (long long)zo * sCo + (long long)zi * sCi;
  #pragma unroll
  for (int mi = 0; mi < 2; ++mi) {
    #pragma unroll
    for (int ni = 0; ni < 2; ++ni) {
      const int col  = n0 + wn * 32 + ni * 16 + mrow;
      const int rowb = m0 + wm * 32 + mi * 16 + (khalf ? 8 : 0);
      #pragma unroll
      for (int r = 0; r < 8; ++r) {
        float v = acc[mi][ni][r] * alpha;
        if (BIAS) v += bias[col];
        const long long off = cbase + (long long)(rowb + r) * ldc + col;
        if (OUTBF16) ((unsigned short*)Cout)[off] = f2bf(v);
        else         ((float*)Cout)[off] = v;
      }
    }
  }
}

// -------------------------------------------------------- conversion kernels
__global__ void cvt_f32_bf16(const float* __restrict__ src,
                             unsigned short* __restrict__ dst, long long n) {
  long long i = (long long)blockIdx.x * blockDim.x + threadIdx.x;
  const long long stride = (long long)gridDim.x * blockDim.x;
  for (; i < n; i += stride) dst[i] = f2bf(src[i]);
}

// E[4096][256] -> Et_bf[256][4096]
__global__ void cvtT_E_bf16(const float* __restrict__ E,
                            unsigned short* __restrict__ dst) {
  long long i = (long long)blockIdx.x * blockDim.x + threadIdx.x;
  if (i < 4096LL * 256LL) {
    const int n  = (int)(i >> 8);
    const int kk = (int)(i & 255);
    dst[(long long)kk * 4096 + n] = f2bf(E[i]);
  }
}

// ------------------------------------------------- wave-per-row softmax(256)
__global__ __launch_bounds__(256)
void softmax_rows256(const float* __restrict__ S,
                     unsigned short* __restrict__ Aout, long long rows) {
  const long long row = (long long)blockIdx.x * 8 + (threadIdx.x >> 5);
  if (row >= rows) return;
  const int lane = threadIdx.x & 31;
  const float* sp = S + row * 256;
  float v[8];
  float m = -3.4e38f;
  #pragma unroll
  for (int j = 0; j < 8; ++j) { v[j] = sp[j * 32 + lane]; m = fmaxf(m, v[j]); }
  #pragma unroll
  for (int s = 16; s >= 1; s >>= 1) m = fmaxf(m, __shfl_xor(m, s, 32));
  float sum = 0.0f;
  #pragma unroll
  for (int j = 0; j < 8; ++j) { v[j] = __expf(v[j] - m); sum += v[j]; }
  #pragma unroll
  for (int s = 16; s >= 1; s >>= 1) sum += __shfl_xor(sum, s, 32);
  const float inv = 1.0f / sum;
  unsigned short* op = Aout + row * 256;
  #pragma unroll
  for (int j = 0; j < 8; ++j) op[j * 32 + lane] = f2bf(v[j] * inv);
}

// -------------------------------------------------------------------- launch
extern "C" void kernel_launch(void* const* d_in, const int* in_sizes, int n_in,
                              void* d_out, int out_size, void* d_ws, size_t ws_size,
                              hipStream_t stream) {
  const float* x    = (const float*)d_in[0];   // [4,4096,1024]
  const float* Wqkv = (const float*)d_in[1];   // [1024,3072]
  const float* E    = (const float*)d_in[2];   // [4096,256]
  const float* Wout = (const float*)d_in[3];   // [1024,1024]
  const float* bout = (const float*)d_in[4];   // [1024]
  float* out = (float*)d_out;

  const long long BN_ = 4LL * 4096;            // B*N = 16384 rows
  const long long D   = 1024, K3 = 3072, KR = 256, HD = 64, H = 16, N = 4096;

  char* ws = (char*)d_ws;
  unsigned short* x_bf    = (unsigned short*)(ws);                    // 32 MB
  unsigned short* wqkv_bf = (unsigned short*)(ws + 33554432LL);       // 6 MB
  unsigned short* et_bf   = (unsigned short*)(ws + 39845888LL);       // 2 MB
  unsigned short* wout_bf = (unsigned short*)(ws + 41943040LL);       // 2 MB
  unsigned short* qkv_bf  = (unsigned short*)(ws + 44040192LL);       // 96 MB
  unsigned short* kp_bf   = (unsigned short*)(ws + 144703488LL);      // 0.5 MB
  unsigned short* vp_bf   = (unsigned short*)(ws + 145227776LL);      // 0.5 MB
  unsigned short* ao_bf   = (unsigned short*)(ws + 145752064LL);      // 32 MB
  unsigned short* attn_bf = (unsigned short*)(ws + 179306496LL);      // 32 MB
  float*          scores  = (float*)         (ws + 212860928LL);      // 64 MB

  // -- fp32 -> bf16 staging
  cvt_f32_bf16<<<4096, 256, 0, stream>>>(x,    x_bf,    BN_ * D);
  cvt_f32_bf16<<<2048, 256, 0, stream>>>(Wqkv, wqkv_bf, D * K3);
  cvt_f32_bf16<<<1024, 256, 0, stream>>>(Wout, wout_bf, D * D);
  cvtT_E_bf16 <<<(int)((N * KR) / 256), 256, 0, stream>>>(E, et_bf);

  // -- 1) qkv = x @ W_qkv : [16384,1024]x[1024,3072] -> bf16 [16384,3072]
  {
    dim3 g(K3 / TILE_N, BN_ / TILE_M, 1);
    gemm_bf16_wmma<false, true, false><<<g, 256, 0, stream>>>(
        x_bf, wqkv_bf, qkv_bf, (int)BN_, (int)K3, (int)D,
        (int)D, (int)K3, (int)K3, 1,
        0, 0, 0, 0, 0, 0, 1.0f, nullptr);
  }

  // -- 2) k_proj / v_proj = E^T @ {k,v}[b,h] : [256,4096]x[4096,64], 64 batches
  {
    dim3 g(HD / TILE_N, KR / TILE_M, 64);   // z = b*16+h
    const long long sBo = N * K3, sBi = HD;                 // into qkv (b,h)
    const long long sCo = H * KR * HD, sCi = KR * HD;       // [64][256][64]
    gemm_bf16_wmma<false, true, false><<<g, 256, 0, stream>>>(
        et_bf, qkv_bf + D /*k block*/, kp_bf, (int)KR, (int)HD, (int)N,
        (int)N, (int)K3, (int)HD, (int)H,
        0, 0, sBo, sBi, sCo, sCi, 1.0f, nullptr);
    gemm_bf16_wmma<false, true, false><<<g, 256, 0, stream>>>(
        et_bf, qkv_bf + 2 * D /*v block*/, vp_bf, (int)KR, (int)HD, (int)N,
        (int)N, (int)K3, (int)HD, (int)H,
        0, 0, sBo, sBi, sCo, sCi, 1.0f, nullptr);
  }

  // -- 3) scores = (q @ kp^T) * 1/sqrt(hd) : [4096,64]x[64,256] -> f32, 64 batches
  {
    dim3 g(KR / TILE_N, N / TILE_M, 64);
    gemm_bf16_wmma<true, false, false><<<g, 256, 0, stream>>>(
        qkv_bf /*q block*/, kp_bf, scores, (int)N, (int)KR, (int)HD,
        (int)K3, (int)HD, (int)KR, (int)H,
        N * K3, HD,                         // A: (b,h) into qkv
        H * KR * HD, KR * HD,               // B: kp batches
        (long long)H * N * KR, N * KR,      // C: [64][4096][256]
        0.125f, nullptr);
  }

  // -- 4) softmax over k=256
  softmax_rows256<<<(int)((4LL * H * N) / 8), 256, 0, stream>>>(
      scores, attn_bf, 4LL * H * N);

  // -- 5) head out = attn @ vp : [4096,256]x[256,64] -> bf16 into [B,N,D]
  {
    dim3 g(HD / TILE_N, N / TILE_M, 64);
    gemm_bf16_wmma<false, true, false><<<g, 256, 0, stream>>>(
        attn_bf, vp_bf, ao_bf, (int)N, (int)HD, (int)KR,
        (int)KR, (int)HD, (int)D, (int)H,
        (long long)H * N * KR, N * KR,      // A: attn batches
        H * KR * HD, KR * HD,               // B: vp batches
        N * D, HD,                          // C: b stride / h*64 col offset
        1.0f, nullptr);
  }

  // -- 6) final: out = ao @ W_out + b_out : [16384,1024]x[1024,1024] -> f32
  {
    dim3 g(D / TILE_N, BN_ / TILE_M, 1);
    gemm_bf16_wmma<false, false, true><<<g, 256, 0, stream>>>(
        ao_bf, wout_bf, out, (int)BN_, (int)D, (int)D,
        (int)D, (int)D, (int)D, 1,
        0, 0, 0, 0, 0, 0, 1.0f, bout);
  }
}
